// XLNetViT_11768210391585
// MI455X (gfx1250) — matrix operand order repaired and checked by reading, not compiled, not run on hardware
//
#include <hip/hip_runtime.h>
#include <hip/hip_bf16.h>

typedef __bf16 bh;
typedef __attribute__((__ext_vector_type__(16))) __bf16 v16bf;
typedef __attribute__((__ext_vector_type__(8)))  float  v8f;
typedef long long ll;

// ---- problem constants ----
#define kB    16
#define kIMG  224
#define kP    16
#define kGRID 14          // IMG/P
#define kL    196         // 14*14
#define kD    768
#define kH    12
#define kHD   64
#define kHID  3072
#define kDEPTH 12
#define kNX   196         // L-1 gathered + cls
#define kNT   392         // 2*L
#define kROWS (kB*kNT)    // 6272
#define kSC   196         // score cols (valid)
#define kPKP  224         // probs cols padded to mult of 32

// ==================== WMMA GEMM ====================
// D = act( scale*(A @ B) + bias + res ), A: MxK bf16 row-major, K mult of 32.
// BT=true : B given as W (N x K) row-major  -> computes A @ W^T
// BT=false: B given as (K x N) row-major (rows clamped to Kvalid-1)
// Wave tile: 32x64 (2 M-tiles x 4 N-tiles; B fragments reused across both M-tiles).
// grid: x = ceil(Nv/64) strips, y = ceil(M/256), z = batches (z1=z/zdiv, z2=z%zdiv)
template <bool BT>
__global__ __launch_bounds__(256) void gemm_bf16(
    const bh* __restrict__ A,  ll sA1, ll sA2, int lda,
    const bh* __restrict__ Bm, ll sB1, ll sB2, int ldb,
    const float* __restrict__ bias,
    const float* __restrict__ res, ll sR1, ll sR2, int ldr,
    float* __restrict__ outF, ll sF1, ll sF2, int ldof,
    bh*    __restrict__ outB, ll sOB1, ll sOB2, int ldob,
    int M, int Nvalid, int K, int Kvalid, int zdiv, int act, float scale)
{
  const int z  = blockIdx.z;
  const int z1 = z / zdiv, z2 = z % zdiv;
  A  += z1 * sA1 + z2 * sA2;
  Bm += z1 * sB1 + z2 * sB2;
  if (res)  res  += z1 * sR1  + z2 * sR2;
  if (outF) outF += z1 * sF1  + z2 * sF2;
  if (outB) outB += z1 * sOB1 + z2 * sOB2;

  const int lane = threadIdx.x & 31;
  const int wave = threadIdx.x >> 5;
  const int mt   = blockIdx.y * 8 + wave;
  const int m0   = mt * 32;
  if (m0 >= M) return;                       // wave-uniform
  const int n0  = blockIdx.x * 64;
  const int hi  = lane >> 4;
  const int l16 = lane & 15;

  int mrow0 = m0 + l16;      if (mrow0 > M - 1) mrow0 = M - 1;   // clamped load rows
  int mrow1 = m0 + 16 + l16; if (mrow1 > M - 1) mrow1 = M - 1;
  const bh* __restrict__ Arow0 = A + (ll)mrow0 * lda;
  const bh* __restrict__ Arow1 = A + (ll)mrow1 * lda;

  v8f c[2][4] = {{v8f{}, v8f{}, v8f{}, v8f{}}, {v8f{}, v8f{}, v8f{}, v8f{}}};

  for (int k0 = 0; k0 < K; k0 += 32) {
    // A fragments: lane (m=l16, hi): a[i] <- A[m][k0 + 16*hi + i]
    v16bf a0, a1;
    #pragma unroll
    for (int i = 0; i < 16; ++i) {
      int ka = k0 + 16 * hi + i;
      a0[i] = Arow0[ka];
      a1[i] = Arow1[ka];
    }
    #pragma unroll
    for (int j = 0; j < 4; ++j) {
      int col = n0 + 16 * j + l16;
      if (col > Nvalid - 1) col = Nvalid - 1;            // clamp (stores masked later)
      v16bf b;
      if (BT) {
        const bh* __restrict__ Brow = Bm + (ll)col * ldb + k0 + 16 * hi;
        #pragma unroll
        for (int i = 0; i < 16; ++i) b[i] = Brow[i];     // b[i] = W[col][k0+16hi+i]
      } else {
        #pragma unroll
        for (int i = 0; i < 16; ++i) {
          int kk = k0 + 16 * hi + i;
          if (kk > Kvalid - 1) kk = Kvalid - 1;          // pad rows multiplied by zero probs
          b[i] = Bm[(ll)kk * ldb + col];
        }
      }
      c[0][j] = __builtin_amdgcn_wmma_f32_16x16x32_bf16(
          false, a0, false, b, (short)0, c[0][j], false, false);
      c[1][j] = __builtin_amdgcn_wmma_f32_16x16x32_bf16(
          false, a1, false, b, (short)0, c[1][j], false, false);
    }
  }

  // Epilogue: C[m0 + 16*t + r + 8*hi][n0 + 16*j + l16]
  #pragma unroll
  for (int t = 0; t < 2; ++t) {
    #pragma unroll
    for (int j = 0; j < 4; ++j) {
      const int col = n0 + 16 * j + l16;
      if (col >= Nvalid) continue;
      const float bv = bias ? bias[col] : 0.0f;
      #pragma unroll
      for (int r = 0; r < 8; ++r) {
        const int row = m0 + 16 * t + r + 8 * hi;
        if (row >= M) continue;
        float acc = c[t][j][r] * scale + bv;
        if (res) acc += res[(ll)row * ldr + col];
        if (act == 1) acc = 0.5f * acc * (1.0f + erff(acc * 0.70710678118f));  // exact GELU
        if (outF) outF[(ll)row * ldof + col] = acc;
        if (outB) outB[(ll)row * ldob + col] = (bh)acc;
      }
    }
  }
}

// ==================== helpers ====================
__device__ __forceinline__ float blockReduceSum(float v, float* sh) {
  #pragma unroll
  for (int o = 16; o > 0; o >>= 1) v += __shfl_xor(v, o, 32);
  const int lane = threadIdx.x & 31, wv = threadIdx.x >> 5;
  __syncthreads();
  if (lane == 0) sh[wv] = v;
  __syncthreads();
  float t = 0.0f;
  #pragma unroll
  for (int i = 0; i < 8; ++i) t += sh[i];
  return t;
}
__device__ __forceinline__ float blockReduceMax(float v, float* sh) {
  #pragma unroll
  for (int o = 16; o > 0; o >>= 1) v = fmaxf(v, __shfl_xor(v, o, 32));
  const int lane = threadIdx.x & 31, wv = threadIdx.x >> 5;
  __syncthreads();
  if (lane == 0) sh[wv] = v;
  __syncthreads();
  float t = -3.4e38f;
  #pragma unroll
  for (int i = 0; i < 8; ++i) t = fmaxf(t, sh[i]);
  return t;
}

// ==================== support kernels ====================
__global__ __launch_bounds__(256) void patchify_kernel(
    const float* __restrict__ imgs, float* __restrict__ patches, bh* __restrict__ patches_bf)
{
  ll idx = (ll)blockIdx.x * blockDim.x + threadIdx.x;
  if (idx >= (ll)kB * kL * kD) return;
  int d = idx % kD;
  int l = (idx / kD) % kL;
  int b = idx / ((ll)kD * kL);
  int c  = d % 3;
  int px = (d / 3) % kP;
  int py = d / (3 * kP);
  int ww = l % kGRID, hh = l / kGRID;
  float v = imgs[(((ll)b * 3 + c) * kIMG + (hh * kP + py)) * kIMG + (ww * kP + px)];
  patches[idx]    = v;
  patches_bf[idx] = (bh)v;
}

__global__ __launch_bounds__(256) void argsort_kernel(
    const float* __restrict__ noise, int* __restrict__ ids)
{
  const int b = blockIdx.x, i = threadIdx.x;
  if (i >= kL) return;
  const float vi = noise[b * kL + i];
  int rank = 0;
  for (int j = 0; j < kL; ++j) {
    float vj = noise[b * kL + j];
    rank += (vj < vi) || (vj == vi && j < i);     // stable argsort
  }
  ids[b * kL + rank] = i;
}

__global__ __launch_bounds__(256) void build_xg_kernel(
    const float* __restrict__ xemb, const int* __restrict__ ids,
    const float* __restrict__ cls, const float* __restrict__ pos,
    const float* __restrict__ maskt, float* __restrict__ xg)
{
  ll idx = (ll)blockIdx.x * blockDim.x + threadIdx.x;
  if (idx >= (ll)kB * kNT * kD) return;
  int d = idx % kD;
  int t = (idx / kD) % kNT;
  int b = idx / ((ll)kD * kNT);
  float v;
  if (t == 0) {
    v = cls[d] + pos[d];                                            // cls + pos[0]
  } else if (t < kNX) {
    int src = ids[b * kL + (t - 1)];
    v = xemb[((ll)b * kL + src) * kD + d] + pos[(src + 1) * kD + d];
  } else {
    int src = ids[b * kL + (t - kNX)];
    v = pos[(src + 1) * kD + d] + maskt[d];
  }
  xg[((ll)b * kNT + t) * kD + d] = v;
}

// One block per output row; 768 cols, 3 per thread.
__global__ __launch_bounds__(256) void ln_kernel(
    const float* __restrict__ x, const float* __restrict__ w, const float* __restrict__ bb,
    bh* __restrict__ y, int rows_in_per_b, int rows_out_per_b, int skip)
{
  __shared__ float sh[8];
  const int ro = blockIdx.x;
  const int b = ro / rows_out_per_b, r = ro % rows_out_per_b;
  const float* xr = x + ((ll)b * rows_in_per_b + skip + r) * kD;
  float v[3], s = 0.0f;
  #pragma unroll
  for (int i = 0; i < 3; ++i) { v[i] = xr[threadIdx.x + 256 * i]; s += v[i]; }
  const float mean = blockReduceSum(s, sh) * (1.0f / kD);
  float s2 = 0.0f;
  #pragma unroll
  for (int i = 0; i < 3; ++i) { float dlt = v[i] - mean; s2 += dlt * dlt; }
  const float var = blockReduceSum(s2, sh) * (1.0f / kD);
  const float rs  = rsqrtf(var + 1e-5f);
  bh* yr = y + (ll)ro * kD;
  #pragma unroll
  for (int i = 0; i < 3; ++i) {
    int c = threadIdx.x + 256 * i;
    yr[c] = (bh)((v[i] - mean) * rs * w[c] + bb[c]);
  }
}

// one block per (b,h,query-row): 196 valid scores -> 224 padded bf16 probs
__global__ __launch_bounds__(256) void softmax_kernel(
    const float* __restrict__ scores, bh* __restrict__ probs)
{
  __shared__ float sh[8];
  const ll bs = (ll)blockIdx.x * kSC;
  const ll bp = (ll)blockIdx.x * kPKP;
  const int t = threadIdx.x;
  float x = (t < kSC) ? scores[bs + t] : -3.4e38f;
  const float mx = blockReduceMax(x, sh);
  float e = (t < kSC) ? __expf(x - mx) : 0.0f;
  const float sum = blockReduceSum(e, sh);
  if (t < kSC)          probs[bp + t] = (bh)(e / sum);
  else if (t < kPKP)    probs[bp + t] = (bh)0.0f;
}

__global__ __launch_bounds__(256) void f2b_kernel(const float* __restrict__ x, bh* __restrict__ y, ll n) {
  for (ll i = (ll)blockIdx.x * blockDim.x + threadIdx.x; i < n; i += (ll)gridDim.x * blockDim.x)
    y[i] = (bh)x[i];
}

__global__ void zero_kernel(float* o) { if (threadIdx.x == 0 && blockIdx.x == 0) o[0] = 0.0f; }

__global__ __launch_bounds__(256) void loss_kernel(
    const float* __restrict__ pred, const float* __restrict__ patches,
    const int* __restrict__ ids, float* __restrict__ out)
{
  __shared__ float sh[8];
  const ll total = (ll)kB * kL * kD;
  float local = 0.0f;
  for (ll idx = (ll)blockIdx.x * blockDim.x + threadIdx.x; idx < total;
       idx += (ll)gridDim.x * blockDim.x) {
    int d = idx % kD;
    int j = (idx / kD) % kL;
    int b = idx / ((ll)kD * kL);
    int src = ids[b * kL + j];
    float diff = pred[((ll)b * kL + j) * kD + d] - patches[((ll)b * kL + src) * kD + d];
    local += diff * diff;
  }
  float bsum = blockReduceSum(local, sh);
  if (threadIdx.x == 0) atomicAdd(out, bsum * (1.0f / (float)((ll)kB * kL * kD)));
}

// ==================== host ====================
static inline void launch_gemm(hipStream_t st, bool bt,
    const bh* A, ll sA1, ll sA2, int lda,
    const bh* Bm, ll sB1, ll sB2, int ldb,
    const float* bias,
    const float* res, ll sR1, ll sR2, int ldr,
    float* oF, ll sF1, ll sF2, int ldof,
    bh* oB, ll sOB1, ll sOB2, int ldob,
    int M, int Nv, int K, int Kv, int nz, int zdiv, int act, float scale)
{
  dim3 g((Nv + 63) / 64, (M + 255) / 256, nz);
  if (bt)
    gemm_bf16<true><<<g, 256, 0, st>>>(A, sA1, sA2, lda, Bm, sB1, sB2, ldb, bias,
        res, sR1, sR2, ldr, oF, sF1, sF2, ldof, oB, sOB1, sOB2, ldob,
        M, Nv, K, Kv, zdiv, act, scale);
  else
    gemm_bf16<false><<<g, 256, 0, st>>>(A, sA1, sA2, lda, Bm, sB1, sB2, ldb, bias,
        res, sR1, sR2, ldr, oF, sF1, sF2, ldof, oB, sOB1, sOB2, ldob,
        M, Nv, K, Kv, zdiv, act, scale);
}

extern "C" void kernel_launch(void* const* d_in, const int* in_sizes, int n_in,
                              void* d_out, int out_size, void* d_ws, size_t ws_size,
                              hipStream_t stream) {
  (void)in_sizes; (void)n_in; (void)out_size; (void)ws_size;
  const float* imgs    = (const float*)d_in[0];
  const float* noise   = (const float*)d_in[1];
  const float* patch_w = (const float*)d_in[2];
  const float* patch_b = (const float*)d_in[3];
  const float* cls_tok = (const float*)d_in[4];
  const float* pos     = (const float*)d_in[5];
  const float* mask_t  = (const float*)d_in[6];
  const float* ln1_w   = (const float*)d_in[7];
  const float* ln1_b   = (const float*)d_in[8];
  const float* qkv_w   = (const float*)d_in[9];
  const float* qkv_b   = (const float*)d_in[10];
  const float* proj_w  = (const float*)d_in[11];
  const float* proj_b  = (const float*)d_in[12];
  const float* ln2_w   = (const float*)d_in[13];
  const float* ln2_b   = (const float*)d_in[14];
  const float* fc1_w   = (const float*)d_in[15];
  const float* fc1_b   = (const float*)d_in[16];
  const float* fc2_w   = (const float*)d_in[17];
  const float* fc2_b   = (const float*)d_in[18];
  const float* norm_w  = (const float*)d_in[19];
  const float* norm_b  = (const float*)d_in[20];
  const float* head_w  = (const float*)d_in[21];
  const float* head_b  = (const float*)d_in[22];
  float* loss_out = (float*)d_out;

  // ---- workspace carve-up ----
  char* cur = (char*)d_ws;
  auto carve = [&](size_t bytes) { void* p = cur; cur += (bytes + 255) & ~(size_t)255; return p; };
  const ll nPatch = (ll)kB * kL * kD;      // 2,408,448
  const ll nRowsD = (ll)kROWS * kD;        // 6272*768
  float* patches    = (float*)carve(nPatch * 4);
  bh*    patches_bf = (bh*)   carve(nPatch * 2);
  int*   ids        = (int*)  carve((ll)kB * kL * 4);
  float* xemb       = (float*)carve(nPatch * 4);
  float* xg         = (float*)carve(nRowsD * 4);
  bh*    hbf        = (bh*)   carve(nRowsD * 2);
  bh*    qkvbf      = (bh*)   carve((ll)kROWS * 3 * kD * 2);
  float* scores     = (float*)carve((ll)kB * kH * kNT * kSC * 4);
  bh*    probs      = (bh*)   carve((ll)kB * kH * kNT * kPKP * 2);
  bh*    obf        = (bh*)   carve(nRowsD * 2);
  bh*    hidbf      = (bh*)   carve((ll)kROWS * kHID * 2);
  bh*    gnorm      = (bh*)   carve(nPatch * 2);
  float* pred       = (float*)carve(nPatch * 4);
  bh*    wbf        = (bh*)   carve((ll)kHID * kD * 2);   // reused per-layer weight buffer

  const int cvtBlocks = 2048;

  // ---- stem ----
  patchify_kernel<<<(unsigned)((nPatch + 255) / 256), 256, 0, stream>>>(imgs, patches, patches_bf);
  argsort_kernel<<<kB, 256, 0, stream>>>(noise, ids);
  f2b_kernel<<<cvtBlocks, 256, 0, stream>>>(patch_w, wbf, (ll)kD * kD);
  launch_gemm(stream, true, patches_bf, 0, 0, kD, wbf, 0, 0, kD, patch_b,
              nullptr, 0, 0, 0, xemb, 0, 0, kD, nullptr, 0, 0, 0,
              kB * kL, kD, kD, kD, 1, 1, 0, 1.0f);
  build_xg_kernel<<<(unsigned)(((ll)kB * kNT * kD + 255) / 256), 256, 0, stream>>>(
      xemb, ids, cls_tok, pos, mask_t, xg);

  // ---- transformer layers ----
  for (int ld = 0; ld < kDEPTH; ++ld) {
    // LN1 -> hbf
    ln_kernel<<<kROWS, 256, 0, stream>>>(xg, ln1_w + ld * kD, ln1_b + ld * kD, hbf, kNT, kNT, 0);
    // QKV
    f2b_kernel<<<cvtBlocks, 256, 0, stream>>>(qkv_w + (ll)ld * 3 * kD * kD, wbf, (ll)3 * kD * kD);
    launch_gemm(stream, true, hbf, 0, 0, kD, wbf, 0, 0, kD, qkv_b + ld * 3 * kD,
                nullptr, 0, 0, 0, nullptr, 0, 0, 0, qkvbf, 0, 0, 3 * kD,
                kROWS, 3 * kD, kD, kD, 1, 1, 0, 1.0f);
    // scores = (q @ k^T) * 1/8, batched over z = b*H + h
    launch_gemm(stream, true,
                qkvbf,            (ll)kNT * 3 * kD, 64, 3 * kD,    // q
                qkvbf + kD,       (ll)kNT * 3 * kD, 64, 3 * kD,    // k as W(N,K)
                nullptr, nullptr, 0, 0, 0,
                scores, (ll)kH * kNT * kSC, (ll)kNT * kSC, kSC,
                nullptr, 0, 0, 0,
                kNT, kSC, kHD, kHD, kB * kH, kH, 0, 0.125f);
    softmax_kernel<<<kB * kH * kNT, 256, 0, stream>>>(scores, probs);
    // o = probs @ v   (B row-major, K padded 224, valid 196)
    launch_gemm(stream, false,
                probs, (ll)kH * kNT * kPKP, (ll)kNT * kPKP, kPKP,
                qkvbf + 2 * kD, (ll)kNT * 3 * kD, 64, 3 * kD,      // v rows (K x 64)
                nullptr, nullptr, 0, 0, 0,
                nullptr, 0, 0, 0,
                obf, (ll)kNT * kD, 64, kD,
                kNT, kHD, kPKP, kSC, kB * kH, kH, 0, 1.0f);
    // proj + residual (in-place f32 carry)
    f2b_kernel<<<cvtBlocks, 256, 0, stream>>>(proj_w + (ll)ld * kD * kD, wbf, (ll)kD * kD);
    launch_gemm(stream, true, obf, 0, 0, kD, wbf, 0, 0, kD, proj_b + ld * kD,
                xg, 0, 0, kD, xg, 0, 0, kD, nullptr, 0, 0, 0,
                kROWS, kD, kD, kD, 1, 1, 0, 1.0f);
    // LN2 -> hbf
    ln_kernel<<<kROWS, 256, 0, stream>>>(xg, ln2_w + ld * kD, ln2_b + ld * kD, hbf, kNT, kNT, 0);
    // FC1 + exact GELU -> hidbf
    f2b_kernel<<<cvtBlocks, 256, 0, stream>>>(fc1_w + (ll)ld * kHID * kD, wbf, (ll)kHID * kD);
    launch_gemm(stream, true, hbf, 0, 0, kD, wbf, 0, 0, kD, fc1_b + ld * kHID,
                nullptr, 0, 0, 0, nullptr, 0, 0, 0, hidbf, 0, 0, kHID,
                kROWS, kHID, kD, kD, 1, 1, 1, 1.0f);
    // FC2 + residual
    f2b_kernel<<<cvtBlocks, 256, 0, stream>>>(fc2_w + (ll)ld * kD * kHID, wbf, (ll)kD * kHID);
    launch_gemm(stream, true, hidbf, 0, 0, kHID, wbf, 0, 0, kHID, fc2_b + ld * kD,
                xg, 0, 0, kD, xg, 0, 0, kD, nullptr, 0, 0, 0,
                kROWS, kD, kHID, kHID, 1, 1, 0, 1.0f);
  }

  // ---- head: final LN over g tokens, projection, MSE ----
  ln_kernel<<<kB * kL, 256, 0, stream>>>(xg, norm_w, norm_b, gnorm, kNT, kL, kNX);
  f2b_kernel<<<cvtBlocks, 256, 0, stream>>>(head_w, wbf, (ll)kD * kD);
  launch_gemm(stream, true, gnorm, 0, 0, kD, wbf, 0, 0, kD, head_b,
              nullptr, 0, 0, 0, pred, 0, 0, kD, nullptr, 0, 0, 0,
              kB * kL, kD, kD, kD, 1, 1, 0, 1.0f);
  zero_kernel<<<1, 32, 0, stream>>>(loss_out);
  loss_kernel<<<4096, 256, 0, stream>>>(pred, patches, ids, loss_out);
}
